// TreeNode_33595234189890
// MI455X (gfx1250) — compile-verified, matrix operand order checked by reading, not run on hardware
//
#include <hip/hip_runtime.h>
#include <hip/hip_bf16.h>

typedef _Float16 f16;
typedef __attribute__((ext_vector_type(16))) _Float16 v16h;
typedef __attribute__((ext_vector_type(8)))  _Float16 v8h;
typedef __attribute__((ext_vector_type(8)))  float    v8f;

namespace {
constexpr int Hd = 1024;          // hidden size
constexpr int Bd = 64;            // batch
constexpr int Sd = 512;           // sequence
constexpr int KT_H = Hd / 32;     // 32 K-tiles of depth 32 over H

#define WMMA_F16(acc, a, b) \
  (acc) = __builtin_amdgcn_wmma_f32_16x16x32_f16(false, (a), false, (b), (short)0, (acc), false, false)

// A operand (16x32 f16 tile) directly from a row-major f16 matrix.
// Lane layout (ISA 7.12.2): lanes 0-15 hold M=lane, K = {0..7, 16..23};
// lanes 16-31 hold M=lane-16, K = {8..15, 24..31}.
__device__ __forceinline__ v16h load_a_tile(const f16* __restrict__ A, int lda,
                                            int m0, int kt, int lane) {
  const f16* p = A + (size_t)(m0 + (lane & 15)) * lda + (kt << 5) + ((lane >> 4) << 3);
  union { v16h v; v8h h[2]; } u;
  u.h[0] = *(const v8h*)(p);        // K 0..7   (or 8..15)
  u.h[1] = *(const v8h*)(p + 16);   // K 16..23 (or 24..31)
  return u.v;
}

// B operand (32x16 f16 tile) from the pre-packed weight buffer:
// tile = (nt*KT + kt), 512 halves, stored exactly in per-lane order.
__device__ __forceinline__ v16h load_b_tile(const f16* __restrict__ P, int KT,
                                            int nt, int kt, int lane) {
  return *(const v16h*)(P + (((size_t)nt * KT + kt) << 9) + (lane << 4));
}

// Double-buffered 16x64 GEMM body: 4 accumulators, A tile shared, next-kt
// tiles fetched before current WMMAs so L2 latency overlaps the matrix ops.
__device__ __forceinline__ void gemm4_db(const f16* __restrict__ A, int lda, int m0,
                                         const f16* __restrict__ P, int nt0, int ntStride,
                                         int lane, v8f& c0, v8f& c1, v8f& c2, v8f& c3) {
  v16h a  = load_a_tile(A, lda, m0, 0, lane);
  v16h b0 = load_b_tile(P, KT_H, nt0 + 0 * ntStride, 0, lane);
  v16h b1 = load_b_tile(P, KT_H, nt0 + 1 * ntStride, 0, lane);
  v16h b2 = load_b_tile(P, KT_H, nt0 + 2 * ntStride, 0, lane);
  v16h b3 = load_b_tile(P, KT_H, nt0 + 3 * ntStride, 0, lane);
  for (int kt = 0; kt + 1 < KT_H; ++kt) {
    v16h an  = load_a_tile(A, lda, m0, kt + 1, lane);
    v16h b0n = load_b_tile(P, KT_H, nt0 + 0 * ntStride, kt + 1, lane);
    v16h b1n = load_b_tile(P, KT_H, nt0 + 1 * ntStride, kt + 1, lane);
    v16h b2n = load_b_tile(P, KT_H, nt0 + 2 * ntStride, kt + 1, lane);
    v16h b3n = load_b_tile(P, KT_H, nt0 + 3 * ntStride, kt + 1, lane);
    WMMA_F16(c0, a, b0); WMMA_F16(c1, a, b1); WMMA_F16(c2, a, b2); WMMA_F16(c3, a, b3);
    a = an; b0 = b0n; b1 = b1n; b2 = b2n; b3 = b3n;
  }
  WMMA_F16(c0, a, b0); WMMA_F16(c1, a, b1); WMMA_F16(c2, a, b2); WMMA_F16(c3, a, b3);
}

// Double-buffered 16x16 GEMM body: single accumulator (short critical path for
// the latency-bound recurrent kernels). Accumulates into `c` (callable twice).
__device__ __forceinline__ void gemm1_db(const f16* __restrict__ A, int lda, int m0,
                                         const f16* __restrict__ P, int nt,
                                         int lane, v8f& c) {
  v16h a = load_a_tile(A, lda, m0, 0, lane);
  v16h b = load_b_tile(P, KT_H, nt, 0, lane);
  for (int kt = 0; kt + 1 < KT_H; ++kt) {
    v16h an = load_a_tile(A, lda, m0, kt + 1, lane);
    v16h bn = load_b_tile(P, KT_H, nt, kt + 1, lane);
    WMMA_F16(c, a, b);
    a = an; b = bn;
  }
  WMMA_F16(c, a, b);
}

__device__ __forceinline__ float sigmoidf_(float x) { return 1.0f / (1.0f + __expf(-x)); }
__device__ __forceinline__ float tanhf_(float x) {
  float e = __expf(-2.0f * x);
  return (1.0f - e) / (1.0f + e);
}
} // namespace

// ---- Pack W[N][K] fp32 (row-major, used as out = A @ W^T) into WMMA B-operand
// ---- f16 layout: B[k][n] = W[n][k], tiles of 32(K) x 16(N), per-lane contiguous.
__global__ void k_pack_b(const float* __restrict__ W, f16* __restrict__ P, int N, int K) {
  int idx = blockIdx.x * 256 + threadIdx.x;
  if (idx >= N * K) return;
  int e    = idx & 15;          // element within lane
  int lane = (idx >> 4) & 31;
  int tile = idx >> 9;
  int KT = K >> 5;
  int kt = tile % KT, nt = tile / KT;
  int n   = (nt << 4) + (lane & 15);
  int grp = lane >> 4;
  int klocal = (e < 8) ? ((grp << 3) + e) : (16 + (grp << 3) + (e - 8));
  P[idx] = (f16)W[(size_t)n * K + (kt << 5) + klocal];
}

// ---- Embedding gather -> f16 X[t*B + b][h]
__global__ void k_gather_x(const int* __restrict__ words, const float* __restrict__ embeds,
                           f16* __restrict__ X16) {
  size_t idx = (size_t)blockIdx.x * 256 + threadIdx.x;  // t*B*H + b*H + h
  int h = (int)(idx & (Hd - 1));
  size_t r = idx >> 10;
  int b = (int)(r & (Bd - 1));
  int t = (int)(r >> 6);
  int w = words[b * Sd + t];
  X16[idx] = (f16)embeds[(size_t)w * Hd + h];
}

__global__ void k_f32_to_f16(const float* __restrict__ src, f16* __restrict__ dst) {
  int idx = blockIdx.x * 256 + threadIdx.x;
  dst[idx] = (f16)src[idx];
}

// ---- Big parallel GEMM: XW[r][:] = X16[r][:] @ Wx^T + bx, r in [0, S*B)
// One wave per 16x64 output strip; A tile reused across 4 B tiles.
__global__ void k_gemm_xw(const f16* __restrict__ X16, const f16* __restrict__ WxP,
                          const float* __restrict__ bx, float* __restrict__ XW) {
  const int lane = threadIdx.x;
  const int m0  = blockIdx.x << 4;   // row tile
  const int nt0 = blockIdx.y << 2;   // 4 consecutive N tiles
  v8f a0 = {}, a1 = {}, a2 = {}, a3 = {};
  gemm4_db(X16, Hd, m0, WxP, nt0, 1, lane, a0, a1, a2, a3);
  const int col   = (nt0 << 4) + (lane & 15);
  const int rbase = m0 + ((lane >> 4) << 3);
#pragma unroll
  for (int r = 0; r < 8; ++r) {
    float* orow = XW + (size_t)(rbase + r) * Hd;
    orow[col]      = a0[r] + bx[col];
    orow[col + 16] = a1[r] + bx[col + 16];
    orow[col + 32] = a2[r] + bx[col + 32];
    orow[col + 48] = a3[r] + bx[col + 48];
  }
}

// ---- Recurrent step, part 1: hx = XW[t] + h @ Wp^T + bp  (stored f16)
// Latency-shaped: 1 N-tile per wave -> 256 waves x 32-WMMA dependent chain.
__global__ void k_step1(const f16* __restrict__ h16, const f16* __restrict__ WpP,
                        const float* __restrict__ XWt, const float* __restrict__ bp,
                        f16* __restrict__ hx16) {
  const int lane = threadIdx.x;
  const int m0 = blockIdx.x << 4;    // 0..3  -> rows of B=64
  const int nt = blockIdx.y;         // 0..63 -> one 16-col tile
  const int col   = (nt << 4) + (lane & 15);
  const int rbase = m0 + ((lane >> 4) << 3);
  // Overlap the fp32 side-input fetch (HBM) with the WMMA body.
#pragma unroll
  for (int r = 0; r < 8; ++r)
    __builtin_prefetch(XWt + (size_t)(rbase + r) * Hd + col, 0, 1);
  v8f acc = {};
  gemm1_db(h16, Hd, m0, WpP, nt, lane, acc);
#pragma unroll
  for (int r = 0; r < 8; ++r) {
    const size_t off = (size_t)(rbase + r) * Hd + col;
    hx16[off] = (f16)(acc[r] + XWt[off] + bp[col]);
  }
}

// ---- Final-root part 1: hx = h_last @ Wx^T + prev_h @ Wp^T + bx + bp
__global__ void k_final1(const f16* __restrict__ hlast16, const f16* __restrict__ WxP,
                         const f16* __restrict__ ph16,    const f16* __restrict__ WpP,
                         const float* __restrict__ bx, const float* __restrict__ bp,
                         f16* __restrict__ hx16) {
  const int lane = threadIdx.x;
  const int m0 = blockIdx.x << 4;
  const int nt = blockIdx.y;
  v8f acc = {};
  gemm1_db(hlast16, Hd, m0, WxP, nt, lane, acc);
  gemm1_db(ph16,    Hd, m0, WpP, nt, lane, acc);
  const int col   = (nt << 4) + (lane & 15);
  const int rbase = m0 + ((lane >> 4) << 3);
#pragma unroll
  for (int r = 0; r < 8; ++r) {
    const size_t off = (size_t)(rbase + r) * Hd + col;
    hx16[off] = (f16)(acc[r] + bx[col] + bp[col]);
  }
}

// ---- Recurrent step, part 2: g = hx @ Wio^T + bio, gates, state update.
// One wave computes the same 16x16 (row, j) tile for all 4 gates (i,o,f,u),
// sharing the A tile across the four B tiles (N offsets j, H+j, 2H+j, 3H+j).
template <bool FINAL>
__global__ void k_step2(const f16* __restrict__ hx16, const f16* __restrict__ WioP,
                        const float* __restrict__ bio,
                        const float* __restrict__ c_in, float* __restrict__ c_out,
                        f16* __restrict__ h16_out,
                        float* __restrict__ out_hidden, float* __restrict__ out_cell) {
  const int lane = threadIdx.x;
  const int m0 = blockIdx.x << 4;     // 0..3
  const int jt = blockIdx.y;          // 0..63, column tile within H
  const int col   = (jt << 4) + (lane & 15);
  const int rbase = m0 + ((lane >> 4) << 3);
  // Overlap cell-state fetch with the WMMA body.
#pragma unroll
  for (int r = 0; r < 8; ++r)
    __builtin_prefetch(c_in + (size_t)(rbase + r) * Hd + col, 0, 1);
  v8f ai = {}, ao = {}, af = {}, au = {};
  gemm4_db(hx16, Hd, m0, WioP, jt, 64, lane, ai, ao, af, au);
  const float bi_ = bio[col], bo_ = bio[Hd + col], bf_ = bio[2 * Hd + col], bu_ = bio[3 * Hd + col];
#pragma unroll
  for (int r = 0; r < 8; ++r) {
    const size_t off = (size_t)(rbase + r) * Hd + col;
    float iv = sigmoidf_(ai[r] + bi_);
    float ov = sigmoidf_(ao[r] + bo_);
    float fv = sigmoidf_(af[r] + bf_);
    float uv = tanhf_(au[r] + bu_);
    float c2 = iv * uv + fv * c_in[off];
    float h2 = ov * tanhf_(c2);
    if (FINAL) {
      out_hidden[off] = h2;
      out_cell[off]   = c2;
    } else {
      c_out[off]   = c2;
      h16_out[off] = (f16)h2;
    }
  }
}

extern "C" void kernel_launch(void* const* d_in, const int* in_sizes, int n_in,
                              void* d_out, int out_size, void* d_ws, size_t ws_size,
                              hipStream_t stream) {
  (void)in_sizes; (void)n_in; (void)out_size; (void)ws_size;
  const int*   words  = (const int*)  d_in[0];
  const float* prev_h = (const float*)d_in[1];
  const float* prev_c = (const float*)d_in[2];
  const float* embeds = (const float*)d_in[3];
  const float* Wx     = (const float*)d_in[4];
  const float* bx     = (const float*)d_in[5];
  const float* Wp     = (const float*)d_in[6];
  const float* bp     = (const float*)d_in[7];
  const float* Wio    = (const float*)d_in[8];
  const float* bio    = (const float*)d_in[9];
  float* out = (float*)d_out;

  // Workspace carve-up (~205 MB total)
  char* ws = (char*)d_ws;
  f16*   WxP  = (f16*)(ws);                               // 2 MB
  f16*   WpP  = (f16*)(ws + (size_t)2  * 1024 * 1024);    // 2 MB
  f16*   WioP = (f16*)(ws + (size_t)4  * 1024 * 1024);    // 8 MB
  f16*   X16  = (f16*)(ws + (size_t)12 * 1024 * 1024);    // 64 MB
  float* XW   = (float*)(ws + (size_t)76 * 1024 * 1024);  // 128 MB
  char*  st   = ws + (size_t)204 * 1024 * 1024;
  f16*   h16  = (f16*)(st);                               // 128 KB
  float* cbuf = (float*)(st + 128 * 1024);                // 256 KB
  f16*   hx16 = (f16*)(st + 384 * 1024);                  // 128 KB
  f16*   ph16 = (f16*)(st + 512 * 1024);                  // 128 KB

  // --- one-time (per call) prep: pack weights, gather embeds, init state ---
  k_pack_b<<<(Hd * Hd) / 256, 256, 0, stream>>>(Wx, WxP, Hd, Hd);
  k_pack_b<<<(Hd * Hd) / 256, 256, 0, stream>>>(Wp, WpP, Hd, Hd);
  k_pack_b<<<(4 * Hd * Hd) / 256, 256, 0, stream>>>(Wio, WioP, 4 * Hd, Hd);
  k_gather_x<<<(size_t)Sd * Bd * Hd / 256, 256, 0, stream>>>(words, embeds, X16);
  k_f32_to_f16<<<(Bd * Hd) / 256, 256, 0, stream>>>(prev_h, ph16);
  hipMemsetAsync(h16, 0, (size_t)Bd * Hd * sizeof(f16), stream);
  hipMemsetAsync(cbuf, 0, (size_t)Bd * Hd * sizeof(float), stream);

  // --- parallel precompute: XW = X @ Wx^T + bx over all S*B rows ---
  k_gemm_xw<<<dim3((Sd * Bd) / 16, Hd / 64), 32, 0, stream>>>(X16, WxP, bx, XW);

  // --- sequential recurrence: 512 steps, 2 small WMMA kernels each ---
  for (int t = 0; t < Sd; ++t) {
    k_step1<<<dim3(Bd / 16, Hd / 16), 32, 0, stream>>>(
        h16, WpP, XW + (size_t)t * Bd * Hd, bp, hx16);
    k_step2<false><<<dim3(Bd / 16, Hd / 16), 32, 0, stream>>>(
        hx16, WioP, bio, cbuf, cbuf, h16, nullptr, nullptr);
  }

  // --- root cell: _cell(h_last, prev_h, prev_c) -> [hidden; cell] ---
  k_final1<<<dim3(Bd / 16, Hd / 16), 32, 0, stream>>>(h16, WxP, ph16, WpP, bx, bp, hx16);
  k_step2<true><<<dim3(Bd / 16, Hd / 16), 32, 0, stream>>>(
      hx16, WioP, bio, prev_c, nullptr, nullptr, out, out + (size_t)Bd * Hd);
}
